// Encoder_22213570855422
// MI455X (gfx1250) — matrix operand order repaired and checked by reading, not compiled, not run on hardware
//
#include <hip/hip_runtime.h>
#include <math.h>

// ---------------------------------------------------------------------------
// Types for CDNA5 WMMA (wave32): D(16x16 f32) = A(16x32 f16) x B(32x16 f16) + C
// ---------------------------------------------------------------------------
typedef _Float16 half_t;
typedef _Float16 v16h __attribute__((ext_vector_type(16)));
typedef float    v8f  __attribute__((ext_vector_type(8)));
typedef unsigned int v4u __attribute__((ext_vector_type(4)));

union FragH {
    v16h h;
    v4u  u[2];
};

__device__ __forceinline__ v8f wmma_f16(v16h a, v16h b, v8f c) {
    // (neg_a, A, neg_b, B, c_mod, C, reuse_a, reuse_b)
    return __builtin_amdgcn_wmma_f32_16x16x32_f16(false, a, false, b, (short)0, c,
                                                  false, false);
}

// Problem constants
#define NB   2
#define NN   1024
#define ND   512
#define NH   8
#define NDH  64
#define NL   4
#define NDFF 2048
#define ROWS (NB * NN)   // 2048

// ---------------------------------------------------------------------------
// Coordinate Fourier features: feats[row, col] = col<256 ? cos(phase) : sin(phase)
// ---------------------------------------------------------------------------
__global__ __launch_bounds__(256) void feats_kernel(const float* __restrict__ coords,
                                                    const float* __restrict__ modes,
                                                    half_t* __restrict__ feats) {
    unsigned idx = blockIdx.x * 256u + threadIdx.x;   // < 2048*512
    unsigned row = idx >> 9;
    unsigned col = idx & 511;
    unsigned m = col & 255;
    const float* c  = coords + row * 3;
    const float* mo = modes + m * 3;
    float ph = c[0] * mo[0] + c[1] * mo[1] + c[2] * mo[2];
    feats[idx] = (half_t)((col < 256) ? cosf(ph) : sinf(ph));
}

// ---------------------------------------------------------------------------
// Pairwise distances (raw, unnormalized) + per-batch sum / sumsq (f64 atomics)
// ---------------------------------------------------------------------------
__global__ __launch_bounds__(256) void dist_kernel(const float* __restrict__ coords,
                                                   float* __restrict__ dmat,
                                                   double* __restrict__ sums) {
    unsigned idx = blockIdx.x * 256u + threadIdx.x;   // < 2 * 1M
    unsigned b = idx >> 20;
    unsigned rem = idx & ((1u << 20) - 1);
    unsigned i = rem >> 10;
    unsigned j = rem & 1023;
    const float* ci = coords + (b * NN + i) * 3;
    const float* cj = coords + (b * NN + j) * 3;
    float dx = ci[0] - (cj[0] + 1e-5f);
    float dy = ci[1] - (cj[1] + 1e-5f);
    float dz = ci[2] - (cj[2] + 1e-5f);
    float d = sqrtf(dx * dx + dy * dy + dz * dz);
    dmat[idx] = d;

    // block reduction of sum, sumsq (whole block shares one b: 1M % 256 == 0)
    float s = d, ss = d * d;
    #pragma unroll
    for (int off = 16; off >= 1; off >>= 1) {
        s  += __shfl_xor(s, off, 32);
        ss += __shfl_xor(ss, off, 32);
    }
    __shared__ float ps[8], pss[8];
    int lane = threadIdx.x & 31, wave = threadIdx.x >> 5;
    if (lane == 0) { ps[wave] = s; pss[wave] = ss; }
    __syncthreads();
    if (wave == 0) {
        float s2  = (lane < 8) ? ps[lane]  : 0.f;
        float ss2 = (lane < 8) ? pss[lane] : 0.f;
        #pragma unroll
        for (int off = 4; off >= 1; off >>= 1) {
            s2  += __shfl_xor(s2, off, 32);
            ss2 += __shfl_xor(ss2, off, 32);
        }
        if (lane == 0) {
            atomicAdd(&sums[b * 2 + 0], (double)s2);
            atomicAdd(&sums[b * 2 + 1], (double)ss2);
        }
    }
}

__global__ void invstd_kernel(const double* __restrict__ sums,
                              float* __restrict__ inv_std) {
    int b = threadIdx.x;
    if (b < NB) {
        double s = sums[b * 2 + 0], ss = sums[b * 2 + 1];
        double n = (double)NN * (double)NN;
        double var = (ss - s * s / n) / (n - 1.0);   // ddof=1
        inv_std[b] = (float)(1.0 / sqrt(var));
    }
}

// ---------------------------------------------------------------------------
// LayerNorm over D=512 per row; 1 wave per row.  OUTF16: write f16, else f32.
// ---------------------------------------------------------------------------
template <bool OUTF16>
__global__ __launch_bounds__(32) void ln_kernel(const float* __restrict__ X,
                                                const float* __restrict__ g,
                                                const float* __restrict__ bb,
                                                float* __restrict__ outF,
                                                half_t* __restrict__ outH) {
    unsigned row = blockIdx.x;
    unsigned lane = threadIdx.x;
    const float* xr = X + row * ND;
    float vals[16];
    float s = 0.f, ss = 0.f;
    #pragma unroll
    for (int i = 0; i < 16; ++i) {
        float v = xr[lane + 32 * i];
        vals[i] = v;
        s += v;
        ss += v * v;
    }
    #pragma unroll
    for (int off = 16; off >= 1; off >>= 1) {
        s  += __shfl_xor(s, off, 32);
        ss += __shfl_xor(ss, off, 32);
    }
    float mean = s * (1.f / ND);
    float var = ss * (1.f / ND) - mean * mean;       // ddof=0
    float rs = rsqrtf(var + 1e-5f);
    #pragma unroll
    for (int i = 0; i < 16; ++i) {
        unsigned c = lane + 32 * i;
        float v = (vals[i] - mean) * rs * g[c] + bb[c];
        if (OUTF16) outH[row * ND + c] = (half_t)v;
        else        outF[row * ND + c] = v;
    }
}

// ---------------------------------------------------------------------------
// Generic WMMA GEMM: C[Rows,NC] = A_f16[Rows,K] @ W_f32[K,NC]
// Block tile 128x128, 8 waves (4x2), each wave 32x64 (2x4 WMMA tiles), K step 32.
// Compile-time epilogue: +bias, +resid, gelu; OUT: 0=f32, 1=f16 row-major,
// 2=f16 V-transposed [B,H,dh,N].
// ---------------------------------------------------------------------------
#define BM 128
#define BN 128
#define LSTR 40   // half elements per LDS row (multiple of 8 -> 16B aligned)

template <bool HAS_BIAS, bool HAS_RESID, bool GELU, int OUT>
__global__ __launch_bounds__(256)
void gemm_kernel(const half_t* __restrict__ A, const float* __restrict__ W,
                 const float* __restrict__ bias, const float* __restrict__ resid,
                 float* __restrict__ outF, half_t* __restrict__ outH,
                 int K, int NC) {
    __shared__ half_t As[BM * LSTR];   // 10 KB
    __shared__ half_t Bs[BN * LSTR];   // 10 KB

    const unsigned tid  = threadIdx.x;
    const unsigned lane = tid & 31;
    const unsigned wave = tid >> 5;
    const unsigned hid  = lane >> 4;        // half-wave id
    const unsigned l16  = lane & 15;
    const unsigned wm   = wave & 3;         // M wave coord (x32)
    const unsigned wn   = wave >> 2;        // N wave coord (x64)
    const unsigned row0 = blockIdx.x * BM;
    const unsigned col0 = blockIdx.y * BN;

    const v8f z8 = {0.f, 0.f, 0.f, 0.f, 0.f, 0.f, 0.f, 0.f};
    v8f acc[2][4] = {{z8, z8, z8, z8}, {z8, z8, z8, z8}};

    const unsigned ldr = tid >> 1;          // 0..127 A-stage row
    const unsigned ldc = (tid & 1) * 16;    // 0 or 16 (half offset)
    const half_t* aSrc = A + (row0 + ldr) * (unsigned)K + ldc;

    for (int kk = 0; kk < K; kk += 32) {
        // stage A tile [128 x 32] f16 (straight copy)
        {
            const v4u* s = (const v4u*)(aSrc + kk);
            v4u* d = (v4u*)(As + ldr * LSTR + ldc);
            d[0] = s[0];
            d[1] = s[1];
        }
        // stage B tile transposed+converted: Bs[n][k] = (f16) W[kk+k][col0+n]
        #pragma unroll
        for (int i = 0; i < 16; ++i) {
            unsigned e = tid + 256 * i;     // 0..4095
            unsigned kr = e >> 7;           // 0..31
            unsigned n  = e & 127;
            Bs[n * LSTR + kr] = (half_t)W[(unsigned)(kk + kr) * (unsigned)NC + col0 + n];
        }
        __syncthreads();

        FragH a[2], bfr[4];
        #pragma unroll
        for (int mt = 0; mt < 2; ++mt) {
            // A frag: lane l: row = l%16, K halves 8*hid..+7 and +16..+23
            const half_t* p = As + (wm * 32 + mt * 16 + l16) * LSTR + hid * 8;
            a[mt].u[0] = *(const v4u*)p;
            a[mt].u[1] = *(const v4u*)(p + 16);
        }
        #pragma unroll
        for (int nt = 0; nt < 4; ++nt) {
            // B frag: lane l: n = l%16, K halves 8*hid..+7 and +16..+23
            const half_t* p = Bs + (wn * 64 + nt * 16 + l16) * LSTR + hid * 8;
            bfr[nt].u[0] = *(const v4u*)p;
            bfr[nt].u[1] = *(const v4u*)(p + 16);
        }
        #pragma unroll
        for (int mt = 0; mt < 2; ++mt)
            #pragma unroll
            for (int nt = 0; nt < 4; ++nt)
                acc[mt][nt] = wmma_f16(a[mt].h, bfr[nt].h, acc[mt][nt]);
        if (kk + 32 < K) __builtin_prefetch(aSrc + kk + 32, 0, 1);
        __syncthreads();
    }

    // Epilogue: C layout M = r + 8*hid, N = l16
    #pragma unroll
    for (int mt = 0; mt < 2; ++mt) {
        #pragma unroll
        for (int nt = 0; nt < 4; ++nt) {
            #pragma unroll
            for (int r = 0; r < 8; ++r) {
                unsigned row = row0 + wm * 32 + mt * 16 + r + 8 * hid;
                unsigned col = col0 + wn * 64 + nt * 16 + l16;
                unsigned off = row * (unsigned)NC + col;
                float v = acc[mt][nt][r];
                if (HAS_BIAS)  v += bias[col];
                if (HAS_RESID) v += resid[off];
                if (GELU)      v = 0.5f * v * (1.f + erff(v * 0.70710678118654752f));
                if (OUT == 0) outF[off] = v;
                if (OUT == 1) outH[off] = (half_t)v;
                if (OUT == 2) {
                    unsigned b = row >> 10, n = row & 1023;
                    unsigned h = col >> 6,  kd = col & 63;
                    outH[((b * NH + h) * NDH + kd) * NN + n] = (half_t)v;
                }
            }
        }
    }
}

// ---------------------------------------------------------------------------
// Flash attention: grid(N/128, B*H), 256 threads; each wave owns 16 queries,
// streams keys in 64-wide tiles with online softmax.  Distance bias:
// score = (q.k)/8 - softplus(temp[h]) * inv_std[b] * d_raw[b,q,key]
// ---------------------------------------------------------------------------
__global__ __launch_bounds__(256)
void attn_kernel(const half_t* __restrict__ Q, const half_t* __restrict__ Kf,
                 const half_t* __restrict__ VT, const float* __restrict__ D,
                 const float* __restrict__ inv_std, const float* __restrict__ temp,
                 half_t* __restrict__ O) {
    __shared__ half_t stage[8 * 16 * 72];   // per-wave 16x64 P tile, stride 72

    const unsigned tid  = threadIdx.x;
    const unsigned lane = tid & 31;
    const unsigned wave = tid >> 5;
    const unsigned hid  = lane >> 4;
    const unsigned l16  = lane & 15;
    const unsigned bh = blockIdx.y;
    const unsigned b  = bh >> 3;
    const unsigned h  = bh & 7;
    const unsigned q0 = blockIdx.x * 128 + wave * 16;

    const float scale = 0.125f;   // 1/sqrt(64)
    float t = temp[h];
    float sp = (t > 0.f) ? (t + log1pf(expf(-t))) : log1pf(expf(t));  // softplus
    const float coeff = sp * inv_std[b];

    // persistent Q A-fragments (K = 0..31, 32..63 of dh)
    FragH qa[2];
    {
        const half_t* qrow = Q + (b * NN + q0 + l16) * ND + h * NDH;
        #pragma unroll
        for (int ks = 0; ks < 2; ++ks) {
            const half_t* p = qrow + ks * 32 + hid * 8;
            qa[ks].u[0] = *(const v4u*)p;
            qa[ks].u[1] = *(const v4u*)(p + 16);
        }
    }

    const v8f z8 = {0.f, 0.f, 0.f, 0.f, 0.f, 0.f, 0.f, 0.f};
    v8f oacc[4] = {z8, z8, z8, z8};
    float mrun[8], lrun[8];
    #pragma unroll
    for (int r = 0; r < 8; ++r) { mrun[r] = -INFINITY; lrun[r] = 0.f; }

    half_t* st = stage + wave * 16 * 72;
    const float* dbase = D + b * NN * NN;

    for (unsigned kb = 0; kb < NN; kb += 64) {
        // ---- S = q @ k^T for 64 keys ----
        v8f s[4] = {z8, z8, z8, z8};
        #pragma unroll
        for (int tn = 0; tn < 4; ++tn) {
            const half_t* krow = Kf + (b * NN + kb + tn * 16 + l16) * ND + h * NDH;
            #pragma unroll
            for (int ks = 0; ks < 2; ++ks) {
                FragH bf;
                const half_t* p = krow + ks * 32 + hid * 8;
                bf.u[0] = *(const v4u*)p;
                bf.u[1] = *(const v4u*)(p + 16);
                s[tn] = wmma_f16(qa[ks].h, bf.h, s[tn]);
            }
        }
        // ---- scale + distance bias + online softmax (per row r) ----
        #pragma unroll
        for (int r = 0; r < 8; ++r) {
            unsigned qrow = q0 + r + 8 * hid;
            const float* drow = dbase + qrow * NN + kb + l16;
            float sv[4];
            #pragma unroll
            for (int tn = 0; tn < 4; ++tn)
                sv[tn] = s[tn][r] * scale - coeff * drow[tn * 16];
            float mx = fmaxf(fmaxf(sv[0], sv[1]), fmaxf(sv[2], sv[3]));
            #pragma unroll
            for (int off = 8; off >= 1; off >>= 1)
                mx = fmaxf(mx, __shfl_xor(mx, off, 16));
            float mnew = fmaxf(mrun[r], mx);
            float corr = __expf(mrun[r] - mnew);
            mrun[r] = mnew;
            float psum = 0.f;
            #pragma unroll
            for (int tn = 0; tn < 4; ++tn) {
                float p = __expf(sv[tn] - mnew);
                psum += p;
                st[(r + 8 * hid) * 72 + tn * 16 + l16] = (half_t)p;
            }
            #pragma unroll
            for (int off = 8; off >= 1; off >>= 1)
                psum += __shfl_xor(psum, off, 16);
            lrun[r] = lrun[r] * corr + psum;
            #pragma unroll
            for (int tn = 0; tn < 4; ++tn) oacc[tn][r] *= corr;
        }
        // ---- O += P @ V  (P from per-wave LDS stage; LDS in-order per wave) ----
        FragH pa[2];
        #pragma unroll
        for (int ks = 0; ks < 2; ++ks) {
            const half_t* p = st + l16 * 72 + ks * 32 + hid * 8;
            pa[ks].u[0] = *(const v4u*)p;
            pa[ks].u[1] = *(const v4u*)(p + 16);
        }
        #pragma unroll
        for (int tn = 0; tn < 4; ++tn) {
            const half_t* vrow = VT + ((b * NH + h) * NDH + tn * 16 + l16) * NN + kb;
            #pragma unroll
            for (int ks = 0; ks < 2; ++ks) {
                FragH bf;
                const half_t* p = vrow + ks * 32 + hid * 8;
                bf.u[0] = *(const v4u*)p;
                bf.u[1] = *(const v4u*)(p + 16);
                oacc[tn] = wmma_f16(pa[ks].h, bf.h, oacc[tn]);
            }
        }
    }
    // ---- write normalized output (f16 row-major [rows, 512]) ----
    #pragma unroll
    for (int tn = 0; tn < 4; ++tn) {
        #pragma unroll
        for (int r = 0; r < 8; ++r) {
            unsigned qrow = q0 + r + 8 * hid;
            float v = oacc[tn][r] / lrun[r];
            O[(b * NN + qrow) * ND + h * NDH + tn * 16 + l16] = (half_t)v;
        }
    }
}

// ---------------------------------------------------------------------------
// Dispatcher
// ---------------------------------------------------------------------------
extern "C" void kernel_launch(void* const* d_in, const int* in_sizes, int n_in,
                              void* d_out, int out_size, void* d_ws, size_t ws_size,
                              hipStream_t stream) {
    const float* phi    = (const float*)d_in[0];
    const float* coords = (const float*)d_in[1];
    const float* modes  = (const float*)d_in[2];
    const float* We     = (const float*)d_in[3];
    const float* be     = (const float*)d_in[4];
    const float* wq     = (const float*)d_in[5];
    const float* bq     = (const float*)d_in[6];
    const float* wk     = (const float*)d_in[7];
    const float* bk     = (const float*)d_in[8];
    const float* wv     = (const float*)d_in[9];
    const float* bv     = (const float*)d_in[10];
    const float* wo     = (const float*)d_in[11];
    const float* bo     = (const float*)d_in[12];
    const float* temp   = (const float*)d_in[13];
    const float* ln1g   = (const float*)d_in[14];
    const float* ln1b   = (const float*)d_in[15];
    const float* ln2g   = (const float*)d_in[16];
    const float* ln2b   = (const float*)d_in[17];
    const float* w1     = (const float*)d_in[18];
    const float* b1     = (const float*)d_in[19];
    const float* w2     = (const float*)d_in[20];
    const float* b2     = (const float*)d_in[21];
    const float* fng    = (const float*)d_in[22];
    const float* fnb    = (const float*)d_in[23];

    char* wsp = (char*)d_ws;
    size_t off = 0;
    auto alloc = [&](size_t bytes) -> void* {
        void* p = wsp + off;
        off += (bytes + 255) & ~(size_t)255;
        return p;
    };
    float*  x      = (float*)alloc((size_t)ROWS * ND * 4);        // residual stream
    half_t* feats  = (half_t*)alloc((size_t)ROWS * ND * 2);
    half_t* y16    = (half_t*)alloc((size_t)ROWS * ND * 2);       // LN1/LN2 out
    half_t* q16    = (half_t*)alloc((size_t)ROWS * ND * 2);
    half_t* k16    = (half_t*)alloc((size_t)ROWS * ND * 2);
    half_t* vT     = (half_t*)alloc((size_t)NB * NH * NDH * NN * 2);
    half_t* o16    = (half_t*)alloc((size_t)ROWS * ND * 2);
    half_t* hbuf   = (half_t*)alloc((size_t)ROWS * NDFF * 2);
    float*  dmat   = (float*)alloc((size_t)NB * NN * NN * 4);
    double* sums   = (double*)alloc(256);
    float*  istd   = (float*)alloc(256);

    hipMemsetAsync(sums, 0, 2 * NB * sizeof(double), stream);

    feats_kernel<<<(ROWS * ND) / 256, 256, 0, stream>>>(coords, modes, feats);
    dist_kernel<<<(NB * NN * NN) / 256, 256, 0, stream>>>(coords, dmat, sums);
    invstd_kernel<<<1, 32, 0, stream>>>(sums, istd);

    dim3 gD(ROWS / BM, ND / BN);        // (16, 4)
    dim3 gF(ROWS / BM, NDFF / BN);      // (16, 16)

    // x = phi + feats @ We + be
    gemm_kernel<true, true, false, 0><<<gD, 256, 0, stream>>>(
        feats, We, be, phi, x, nullptr, ND, ND);

    for (int l = 0; l < NL; ++l) {
        ln_kernel<true><<<ROWS, 32, 0, stream>>>(x, ln1g + l * ND, ln1b + l * ND,
                                                 nullptr, y16);
        gemm_kernel<true, false, false, 1><<<gD, 256, 0, stream>>>(
            y16, wq + (size_t)l * ND * ND, bq + l * ND, nullptr, nullptr, q16,
            ND, ND);
        gemm_kernel<true, false, false, 1><<<gD, 256, 0, stream>>>(
            y16, wk + (size_t)l * ND * ND, bk + l * ND, nullptr, nullptr, k16,
            ND, ND);
        gemm_kernel<true, false, false, 2><<<gD, 256, 0, stream>>>(
            y16, wv + (size_t)l * ND * ND, bv + l * ND, nullptr, nullptr, vT,
            ND, ND);
        attn_kernel<<<dim3(NN / 128, NB * NH), 256, 0, stream>>>(
            q16, k16, vT, dmat, istd, temp + l * NH, o16);
        // x = x + o @ wo + bo   (in-place safe: elementwise resid)
        gemm_kernel<true, true, false, 0><<<gD, 256, 0, stream>>>(
            o16, wo + (size_t)l * ND * ND, bo + l * ND, x, x, nullptr, ND, ND);
        ln_kernel<true><<<ROWS, 32, 0, stream>>>(x, ln2g + l * ND, ln2b + l * ND,
                                                 nullptr, y16);
        // h = gelu(z @ w1 + b1)
        gemm_kernel<true, false, true, 1><<<gF, 256, 0, stream>>>(
            y16, w1 + (size_t)l * ND * NDFF, b1 + l * NDFF, nullptr, nullptr,
            hbuf, ND, NDFF);
        // x = x + h @ w2 + b2
        gemm_kernel<true, true, false, 0><<<gD, 256, 0, stream>>>(
            hbuf, w2 + (size_t)l * NDFF * ND, b2 + l * ND, x, x, nullptr,
            NDFF, ND);
    }
    ln_kernel<false><<<ROWS, 32, 0, stream>>>(x, fng, fnb, (float*)d_out, nullptr);
}
